// ProbTAttention_33818572489252
// MI455X (gfx1250) — matrix operand order repaired and checked by reading, not compiled, not run on hardware
//
#include <hip/hip_runtime.h>
#include <hip/hip_bf16.h>
#include <math.h>
#include <stdint.h>

// Problem constants (match reference)
#define BB   2
#define SS   2048
#define HIDD 2048
#define NHH  32
#define NKVV 8
#define HDD  64
#define GG   4

typedef __bf16 bf16_t;
typedef __attribute__((ext_vector_type(16))) __bf16 v16bf;
typedef __attribute__((ext_vector_type(8)))  float  v8f;
typedef __attribute__((ext_vector_type(4)))  int    v4i;   // clang vector (not HIP_vector_type)

union FragAB {
    v16bf    v;
    unsigned u[8];
    bf16_t   h[16];
    int4     q[2];
};

union B16x16 {
    int4   q[2];
    bf16_t h[16];
};

// ---------------------------------------------------------------------------
// Async global->LDS copy (CDNA5 GLOBAL_LOAD_ASYNC_TO_LDS_B128, ASYNCcnt).
// Builtin signature (from hipcc diagnostics): param0 = v4i* in AS1 (global),
// param1 = v4i* in AS3 (LDS), then two immediate ints (offset, cpol).
// v4i must be the clang vector type, NOT HIP_vector_type<int,4>.
// Guarded: falls back to register-staged copy if the builtin is absent.
// ---------------------------------------------------------------------------
#if __has_builtin(__builtin_amdgcn_global_load_async_to_lds_b128)
#define HAVE_ASYNC_LDS 1
#else
#define HAVE_ASYNC_LDS 0
#endif

__device__ __forceinline__ void async_copy16(const bf16_t* g, bf16_t* l) {
#if HAVE_ASYNC_LDS
    typedef __attribute__((address_space(1))) v4i as1_v4i;
    typedef __attribute__((address_space(3))) v4i as3_v4i;
    __builtin_amdgcn_global_load_async_to_lds_b128(
        (as1_v4i*)(uintptr_t)g,         // global src (AS1; flat == global repr)
        (as3_v4i*)(v4i*)l,              // LDS dst (AS3 via addrspacecast)
        0, 0);
#else
    *(int4*)l = *(const int4*)g;
#endif
}

__device__ __forceinline__ void async_wait0() {
#if HAVE_ASYNC_LDS
#if __has_builtin(__builtin_amdgcn_s_wait_asynccnt)
    __builtin_amdgcn_s_wait_asynccnt(0);
#else
    asm volatile("s_wait_asynccnt 0" ::: "memory");
#endif
#endif
}

// ---------------------------------------------------------------------------
// f32 -> bf16 downconvert (bandwidth pass)
// ---------------------------------------------------------------------------
__global__ void cvt_f32_to_bf16(const float* __restrict__ src,
                                bf16_t* __restrict__ dst, int n) {
    int i = blockIdx.x * blockDim.x + threadIdx.x;
    int stride = gridDim.x * blockDim.x;
    for (; i < n; i += stride) dst[i] = (bf16_t)src[i];
}

// ---------------------------------------------------------------------------
// Tiled bf16 WMMA GEMM: C[M,N] = A[M,K] @ B[K,N]
// Block = 256 thr = 8 waves (4 M-waves x 2 N-waves), block tile 64x128.
// Wave tile 16x64: one A fragment feeds 4 WMMAs per 32-wide K step.
// A tile staged via async-to-LDS (double buffered); B tile staged
// column-major in LDS so B fragments are 2x ds_load_b128.
// One barrier per K step; global B prefetch overlaps the WMMAs.
// ---------------------------------------------------------------------------
template <bool OUT_BF16>
__global__ __launch_bounds__(256) void wmma_gemm_kernel(
    const bf16_t* __restrict__ A, const bf16_t* __restrict__ Bm,
    void* __restrict__ Cout, int M, int N, int K)
{
    __shared__ bf16_t Abuf[2][64][32];    // 8 KB  (row-major)
    __shared__ bf16_t Bbuf[2][128][32];   // 16 KB (column-major: [n][k])

    const int tid  = threadIdx.x;
    const int wid  = tid >> 5;
    const int lane = tid & 31;
    const int h    = lane >> 4;     // lane half (wave32)
    const int ml   = lane & 15;
    const int wm   = wid & 3;       // 0..3 : wave row (16 rows each)
    const int wn   = wid >> 2;      // 0..1 : wave col (64 cols each)
    const int m0   = blockIdx.y * 64;
    const int n0   = blockIdx.x * 128;

    // staging assignments
    const int ra = tid >> 2,  ca = (tid & 3) * 8;    // A: 8 elems (b128)
    const int rb = tid >> 3,  cb = (tid & 7) * 16;   // B: 16 elems

    v8f acc[4] = {};

    // --- prologue: kick off tile kt=0 ---
    async_copy16(&A[(size_t)(m0 + ra) * K + ca], &Abuf[0][ra][ca]);
    B16x16 breg;
    breg.q[0] = *(const int4*)&Bm[(size_t)rb * N + n0 + cb];
    breg.q[1] = *(const int4*)&Bm[(size_t)rb * N + n0 + cb + 8];

    int p = 0;
    for (int kt = 0; kt < K; kt += 32) {
        async_wait0();   // my A-tile async into Abuf[p] is complete
        // scatter B regs column-major into Bbuf[p]
#pragma unroll
        for (int j = 0; j < 16; ++j) Bbuf[p][cb + j][rb] = breg.h[j];
        __syncthreads();

        const bool more = (kt + 32) < K;
        if (more) {  // prefetch next B tile into regs (overlaps compute)
            breg.q[0] = *(const int4*)&Bm[(size_t)(kt + 32 + rb) * N + n0 + cb];
            breg.q[1] = *(const int4*)&Bm[(size_t)(kt + 32 + rb) * N + n0 + cb + 8];
        }

        // A fragment: two contiguous 16B runs of row (wm*16+ml)
        FragAB fa;
        fa.q[0] = *(const int4*)&Abuf[p][wm * 16 + ml][8 * h];
        fa.q[1] = *(const int4*)&Abuf[p][wm * 16 + ml][16 + 8 * h];

#pragma unroll
        for (int j = 0; j < 4; ++j) {
            const int c = wn * 64 + j * 16 + ml;   // this lane's B column
            FragAB fb;
            fb.q[0] = *(const int4*)&Bbuf[p][c][8 * h];
            fb.q[1] = *(const int4*)&Bbuf[p][c][16 + 8 * h];
            acc[j] = __builtin_amdgcn_wmma_f32_16x16x32_bf16(
                false, fa.v, false, fb.v, (short)0, acc[j], false, false);
        }

        if (more)  // safe: reaching here implies all waves passed this barrier
            async_copy16(&A[(size_t)(m0 + ra) * K + kt + 32 + ca],
                         &Abuf[p ^ 1][ra][ca]);
        p ^= 1;
    }

    // C layout: VGPR r -> row r + 8h, col = ml
    const int row0 = m0 + wm * 16 + 8 * h;
#pragma unroll
    for (int j = 0; j < 4; ++j) {
        const int col = n0 + wn * 64 + j * 16 + ml;
#pragma unroll
        for (int r = 0; r < 8; ++r) {
            float val = acc[j][r];
            if (OUT_BF16)
                ((bf16_t*)Cout)[(size_t)(row0 + r) * N + col] = (bf16_t)val;
            else
                ((float*)Cout)[(size_t)(row0 + r) * N + col] = val;
        }
    }
}

// ---------------------------------------------------------------------------
// Flash attention (GQA): one wave per (b, kv-head e, group g, 16-row q tile).
// Scores = Q(16x64) . K^T(64x16) per kv tile via 2 chained K=32 WMMAs;
// online softmax with 16-lane shfl_xor row reductions; P re-layout via LDS;
// P(16x32) . V(32x64) via 4 WMMAs into the running 16x64 output tile.
// All Q/K/P fragment loads are 2x 16-byte reads (contiguous layout runs).
// ---------------------------------------------------------------------------
__global__ __launch_bounds__(256) void flash_attn_kernel(
    const bf16_t* __restrict__ Q, const bf16_t* __restrict__ Kc,
    const bf16_t* __restrict__ Vc, const float* __restrict__ mask,
    bf16_t* __restrict__ Ctx)
{
    __shared__ bf16_t Plds[8][16][32];   // 8 KB, one 16x32 P tile per wave

    const int tid  = threadIdx.x;
    const int wid  = tid >> 5;
    const int lane = tid & 31;
    const int h    = lane >> 4;
    const int ml   = lane & 15;

    int gw = blockIdx.x * 8 + wid;
    const int QT = SS / 16;
    const int qt = gw % QT; gw /= QT;
    const int g  = gw % GG; gw /= GG;
    const int e  = gw % NKVV; gw /= NKVV;
    const int b  = gw;
    const int head = e * GG + g;
    const int q0   = qt * 16;

    // Q fragments for both 32-wide d-halves (HD=64)
    FragAB qf[2];
    {
        const bf16_t* qrow =
            Q + (size_t)(b * SS + q0 + ml) * (NHH * HDD) + head * HDD;
#pragma unroll
        for (int dh = 0; dh < 2; ++dh) {
            qf[dh].q[0] = *(const int4*)&qrow[dh * 32 + 16 * h];
            qf[dh].q[1] = *(const int4*)&qrow[dh * 32 + 16 * h + 8];
        }
    }

    v8f o0 = {}, o1 = {}, o2 = {}, o3 = {};
    float mi[8], li[8];
#pragma unroll
    for (int r = 0; r < 8; ++r) { mi[r] = -1e30f; li[r] = 0.f; }

    const float scale = 0.125f;  // 1/sqrt(64)

    for (int kc = 0; kc < SS; kc += 32) {
        // ---- scores: two 16x16 tiles covering kv columns kc..kc+31 ----
        v8f s0 = {}, s1 = {};
#pragma unroll
        for (int t = 0; t < 2; ++t) {
            const int kvcol = kc + t * 16 + ml;   // this lane's B column
            const bf16_t* krow =
                Kc + (size_t)(b * SS + kvcol) * (NKVV * HDD) + e * HDD;
            FragAB kb0, kb1;
            kb0.q[0] = *(const int4*)&krow[16 * h];        // d 0..31 run
            kb0.q[1] = *(const int4*)&krow[16 * h + 8];
            kb1.q[0] = *(const int4*)&krow[32 + 16 * h];   // d 32..63 run
            kb1.q[1] = *(const int4*)&krow[32 + 16 * h + 8];
            v8f sc = {};
            sc = __builtin_amdgcn_wmma_f32_16x16x32_bf16(
                false, qf[0].v, false, kb0.v, (short)0, sc, false, false);
            sc = __builtin_amdgcn_wmma_f32_16x16x32_bf16(
                false, qf[1].v, false, kb1.v, (short)0, sc, false, false);
            if (t == 0) s0 = sc; else s1 = sc;
        }

        // ---- scale + mask + online softmax (rows r+8h live on this lane) ----
        const float* mbase = mask + (size_t)b * SS * SS;
        const int col0 = kc + ml;
#pragma unroll
        for (int r = 0; r < 8; ++r) {
            const int row = q0 + r + 8 * h;
            const float* mr = mbase + (size_t)row * SS;
            float a = s0[r] * scale + mr[col0];
            float c = s1[r] * scale + mr[col0 + 16];
            float mx = fmaxf(a, c);
#pragma unroll
            for (int d = 1; d < 16; d <<= 1)
                mx = fmaxf(mx, __shfl_xor(mx, d, 32));
            float mnew = fmaxf(mi[r], mx);
            float fac  = __expf(mi[r] - mnew);
            float p0   = __expf(a - mnew);
            float p1   = __expf(c - mnew);
            float rs   = p0 + p1;
#pragma unroll
            for (int d = 1; d < 16; d <<= 1)
                rs += __shfl_xor(rs, d, 32);
            li[r] = li[r] * fac + rs;
            mi[r] = mnew;
            o0[r] *= fac; o1[r] *= fac; o2[r] *= fac; o3[r] *= fac;
            s0[r] = p0; s1[r] = p1;
        }

        // ---- P: C-layout -> row-major LDS -> A-fragment layout ----
        __syncthreads();
#pragma unroll
        for (int r = 0; r < 8; ++r) {
            Plds[wid][r + 8 * h][ml]      = (bf16_t)s0[r];
            Plds[wid][r + 8 * h][16 + ml] = (bf16_t)s1[r];
        }
        __syncthreads();

        FragAB pf;
        pf.q[0] = *(const int4*)&Plds[wid][ml][16 * h];
        pf.q[1] = *(const int4*)&Plds[wid][ml][16 * h + 8];

        // ---- P(16x32) . V(32x64): 4 accumulator tiles over d ----
#pragma unroll
        for (int j = 0; j < 4; ++j) {
            FragAB vf;
            const int dcol = j * 16 + ml;
#pragma unroll
            for (int v = 0; v < 8; ++v) {
                const int kvrow = kc + 16 * h + 2 * v;
                const bf16_t* vb =
                    Vc + (size_t)(b * SS + kvrow) * (NKVV * HDD) + e * HDD + dcol;
                vf.h[2 * v]     = vb[0];
                vf.h[2 * v + 1] = vb[NKVV * HDD];   // next kv row
            }
            v8f acc = (j == 0) ? o0 : (j == 1) ? o1 : (j == 2) ? o2 : o3;
            acc = __builtin_amdgcn_wmma_f32_16x16x32_bf16(
                false, pf.v, false, vf.v, (short)0, acc, false, false);
            if (j == 0) o0 = acc; else if (j == 1) o1 = acc;
            else if (j == 2) o2 = acc; else o3 = acc;
        }
    }

    // ---- epilogue: normalize, write context [b, s, head*64 + d] ----
#pragma unroll
    for (int r = 0; r < 8; ++r) {
        float inv = 1.f / li[r];
        const int row = q0 + r + 8 * h;
        bf16_t* crow = Ctx + (size_t)(b * SS + row) * (NHH * HDD) + head * HDD;
        crow[0 * 16 + ml] = (bf16_t)(o0[r] * inv);
        crow[1 * 16 + ml] = (bf16_t)(o1[r] * inv);
        crow[2 * 16 + ml] = (bf16_t)(o2[r] * inv);
        crow[3 * 16 + ml] = (bf16_t)(o3[r] * inv);
    }
}

// ---------------------------------------------------------------------------
extern "C" void kernel_launch(void* const* d_in, const int* in_sizes, int n_in,
                              void* d_out, int out_size, void* d_ws, size_t ws_size,
                              hipStream_t stream) {
    const float* hs   = (const float*)d_in[0];
    const float* mask = (const float*)d_in[1];
    const float* Wq   = (const float*)d_in[2];
    const float* Wk   = (const float*)d_in[3];
    const float* Wv   = (const float*)d_in[4];
    const float* Wo   = (const float*)d_in[5];
    float*       out  = (float*)d_out;

    // workspace layout (bf16 elements)
    bf16_t* ws = (bf16_t*)d_ws;
    const size_t nX  = (size_t)BB * SS * HIDD;        // 8,388,608
    const size_t nWq = (size_t)HIDD * NHH * HDD;      // 4,194,304
    const size_t nWk = (size_t)HIDD * NKVV * HDD;     // 1,048,576
    const size_t nKV = (size_t)BB * SS * NKVV * HDD;  // 2,097,152
    bf16_t* Xb  = ws;
    bf16_t* Wqb = Xb  + nX;
    bf16_t* Wkb = Wqb + nWq;
    bf16_t* Wvb = Wkb + nWk;
    bf16_t* Wob = Wvb + nWk;
    bf16_t* Qb  = Wob + nWq;
    bf16_t* Kb  = Qb  + nX;
    bf16_t* Vb  = Kb  + nKV;
    bf16_t* Cb  = Vb  + nKV;

    dim3 blk(256);
    cvt_f32_to_bf16<<<2048, blk, 0, stream>>>(hs, Xb,  (int)nX);
    cvt_f32_to_bf16<<<1024, blk, 0, stream>>>(Wq, Wqb, (int)nWq);
    cvt_f32_to_bf16<<<512,  blk, 0, stream>>>(Wk, Wkb, (int)nWk);
    cvt_f32_to_bf16<<<512,  blk, 0, stream>>>(Wv, Wvb, (int)nWk);
    cvt_f32_to_bf16<<<1024, blk, 0, stream>>>(Wo, Wob, (int)nWq);

    const int M = BB * SS;  // 4096
    // Q/K/V projections (bf16 outputs); block tile 64x128
    wmma_gemm_kernel<true><<<dim3(HIDD / 128, M / 64), blk, 0, stream>>>(
        Xb, Wqb, Qb, M, HIDD, HIDD);
    wmma_gemm_kernel<true><<<dim3((NKVV * HDD) / 128, M / 64), blk, 0, stream>>>(
        Xb, Wkb, Kb, M, NKVV * HDD, HIDD);
    wmma_gemm_kernel<true><<<dim3((NKVV * HDD) / 128, M / 64), blk, 0, stream>>>(
        Xb, Wvb, Vb, M, NKVV * HDD, HIDD);

    // attention: B*NKV*G*(S/16) = 8192 waves, 8 waves/block
    flash_attn_kernel<<<(BB * NKVV * GG * (SS / 16)) / 8, blk, 0, stream>>>(
        Qb, Kb, Vb, mask, Cb);

    // output projection -> f32 d_out
    wmma_gemm_kernel<false><<<dim3(HIDD / 128, M / 64), blk, 0, stream>>>(
        Cb, Wob, out, M, HIDD, HIDD);
}
